// Recycling_3307124817989
// MI455X (gfx1250) — compile-verified, matrix operand order checked by reading, not run on hardware
//
#include <hip/hip_runtime.h>
#include <hip/hip_bf16.h>
#include <math.h>

typedef __attribute__((ext_vector_type(16))) _Float16 f16x16;
typedef __attribute__((ext_vector_type(8)))  _Float16 f16x8;
typedef __attribute__((ext_vector_type(8)))  float    f32x8;
typedef __attribute__((ext_vector_type(4)))  float    f32x4;

#define LSEQ 768
#define DPAIR 128
#define DMSA 256
#define DSTATE 32

// ---------------- scratch layout (bytes) ----------------
// [0, 12288)          : Cb as float[768][4]
// [12288, 61440)      : stateH as _Float16[768][32]
// [61440, 94208)      : w fragments as _Float16[8 nt][4 ks][32 lane][16 e]

// =============== Kernel 1: Cb per residue ===============
__global__ void k_cb(const float* __restrict__ xyz, float* __restrict__ Cb) {
    int l = blockIdx.x * blockDim.x + threadIdx.x;
    if (l >= LSEQ) return;
    const float* p = xyz + l * 9;
    float Nx=p[0],Ny=p[1],Nz=p[2], Ax=p[3],Ay=p[4],Az=p[5], Cx=p[6],Cy=p[7],Cz=p[8];
    float bx=Ax-Nx, by=Ay-Ny, bz=Az-Nz;
    float cx=Cx-Ax, cy=Cy-Ay, cz=Cz-Az;
    float ax=by*cz-bz*cy, ay=bz*cx-bx*cz, az=bx*cy-by*cx;
    Cb[l*4+0] = -0.58273431f*ax + 0.56802827f*bx - 0.54067466f*cx + Ax;
    Cb[l*4+1] = -0.58273431f*ay + 0.56802827f*by - 0.54067466f*cy + Ay;
    Cb[l*4+2] = -0.58273431f*az + 0.56802827f*bz - 0.54067466f*cz + Az;
    Cb[l*4+3] = 0.f;
}

// ======= Kernel 2: pack w_dist[128,128] f32 -> f16 WMMA-B fragment order =======
__global__ void k_packw(const float* __restrict__ w, _Float16* __restrict__ wf) {
    int idx = blockIdx.x * blockDim.x + threadIdx.x;   // k*128 + n
    if (idx >= DPAIR * DPAIR) return;
    int k = idx >> 7, n = idx & 127;
    int nt  = n >> 4;
    int ks  = k >> 5;
    int kr  = k & 31;
    int grp = kr >> 3;                 // 0..3
    int lane = (n & 15) | ((grp & 1) << 4);
    int e    = (kr & 7) + ((grp >> 1) << 3);
    wf[((((nt << 2) + ks) << 5) | lane) * 16 + e] = (_Float16)w[idx];
}

// =============== Kernel 3: state LayerNorm (one wave per row) ===============
__global__ void k_state_ln(const float* __restrict__ state,
                           const float* __restrict__ g, const float* __restrict__ b,
                           float* __restrict__ outState, _Float16* __restrict__ stateH) {
    int row = blockIdx.x * 8 + (threadIdx.x >> 5);
    int c = threadIdx.x & 31;
    float x = state[row * DSTATE + c];
    float s = x, ss = x * x;
    #pragma unroll
    for (int m = 16; m >= 1; m >>= 1) {
        s  += __shfl_xor(s,  m, 32);
        ss += __shfl_xor(ss, m, 32);
    }
    float mu = s * (1.f / 32.f);
    float var = ss * (1.f / 32.f) - mu * mu;
    float rs = rsqrtf(var + 1e-5f);
    float y = (x - mu) * rs * g[c] + b[c];
    outState[row * DSTATE + c] = y;
    stateH[row * DSTATE + c] = (_Float16)y;
}

// =============== Kernel 4: msa LayerNorm (256 thr per row) ===============
__global__ void k_msa_ln(const float* __restrict__ msa,
                         const float* __restrict__ g, const float* __restrict__ b,
                         float* __restrict__ out) {
    __shared__ float red[17];
    int row = blockIdx.x, c = threadIdx.x;
    float x = msa[row * DMSA + c];
    float s = x, ss = x * x;
    #pragma unroll
    for (int m = 16; m >= 1; m >>= 1) {
        s  += __shfl_xor(s,  m, 32);
        ss += __shfl_xor(ss, m, 32);
    }
    int wid = c >> 5;
    if ((c & 31) == 0) { red[wid] = s; red[8 + wid] = ss; }
    __syncthreads();
    if (c == 0) {
        float S = 0.f, SS = 0.f;
        #pragma unroll
        for (int w = 0; w < 8; w++) { S += red[w]; SS += red[8 + w]; }
        float mu = S * (1.f / 256.f);
        red[0] = mu;
        red[1] = rsqrtf(SS * (1.f / 256.f) - mu * mu + 1e-5f);
    }
    __syncthreads();
    float mu = red[0], rs = red[1];
    out[row * DMSA + c] = (x - mu) * rs * g[c] + b[c];
}

// =============== Kernel 5: fused dist/rbf/concat -> WMMA -> pair LN + add ===============
// Block = 256 thr (8 waves). Tile = 16 pair-rows (fixed i, 16 consecutive j) x 128 ch.
// pair tensor is touched exactly once (1 coalesced read + 1 coalesced write).
__global__ __launch_bounds__(256)
void k_pair_fused(const float* __restrict__ pair,
                  const float* __restrict__ bdist,
                  const float* __restrict__ gpair, const float* __restrict__ bnpair,
                  const float* __restrict__ Cb, const _Float16* __restrict__ stateH,
                  const _Float16* __restrict__ wfrag,
                  float* __restrict__ outPair) {
    __shared__ __align__(32) _Float16 smA[16 * 136];   // f16 features, 272B/row
    __shared__ __align__(16) float    smP[16 * 132];   // f32 pair tile, 528B/row
    __shared__ float smMu[16], smRs[16];

    int tid = threadIdx.x;
    int i  = blockIdx.x / (LSEQ / 16);
    int j0 = (blockIdx.x % (LSEQ / 16)) * 16;

    // ---- stage phase: 16 threads per row, 8 consecutive channels each ----
    int r   = tid >> 4;        // row in tile
    int sub = tid & 15;
    int j   = j0 + r;
    int c0  = sub * 8;

    float dx = Cb[i*4+0] - Cb[j*4+0];
    float dy = Cb[i*4+1] - Cb[j*4+1];
    float dz = Cb[i*4+2] - Cb[j*4+2];
    float d2 = dx*dx + dy*dy + dz*dz;
    float dist = sqrtf(fmaxf(d2, 1e-12f));

    // coalesced streaming read of pair tile (512B per row), stash in LDS
    const f32x4* prow4 = (const f32x4*)(pair + ((long)(i * LSEQ + j)) * DPAIR + c0);
    f32x4 p0 = __builtin_nontemporal_load(prow4);
    f32x4 p1 = __builtin_nontemporal_load(prow4 + 1);
    *(f32x4*)&smP[r * 132 + c0]     = p0;
    *(f32x4*)&smP[r * 132 + c0 + 4] = p1;

    float s  = p0.x + p0.y + p0.z + p0.w + p1.x + p1.y + p1.z + p1.w;
    float ss = p0.x*p0.x + p0.y*p0.y + p0.z*p0.z + p0.w*p0.w
             + p1.x*p1.x + p1.y*p1.y + p1.z*p1.z + p1.w*p1.w;
    #pragma unroll
    for (int m = 8; m >= 1; m >>= 1) {          // reduce over 16-lane half
        s  += __shfl_xor(s,  m, 32);
        ss += __shfl_xor(ss, m, 32);
    }
    if (sub == 0) {
        float mu = s * (1.f / 128.f);
        float var = ss * (1.f / 128.f) - mu * mu;
        smMu[r] = mu;
        smRs[r] = rsqrtf(var + 1e-5f);
    }

    // features: [0,64)=rbf(dist), [64,96)=stateLN[i], [96,128)=stateLN[j]
    #pragma unroll
    for (int e = 0; e < 8; e++) {
        int c = c0 + e;
        _Float16 f;
        if (c < 64) {
            float mu_k = 2.f + (float)c * (20.f / 63.f);
            float t = (dist - mu_k) * 3.2f;      // 1/sigma, sigma = 20/64
            f = (_Float16)__expf(-t * t);
        } else if (c < 96) {
            f = stateH[i * DSTATE + (c - 64)];
        } else {
            f = stateH[j * DSTATE + (c - 96)];
        }
        smA[r * 136 + c] = f;
    }
    __syncthreads();

    // ---- WMMA phase: wave w owns output cols [w*16, w*16+16) ----
    int wave = tid >> 5;
    int lane = tid & 31;
    int lhi = lane >> 4, llo = lane & 15;

    f32x8 acc = {};
    #pragma unroll
    for (int ks = 0; ks < 4; ks++) {
        int klo = ks * 32 + lhi * 8;
        f16x8 alo = *(const f16x8*)&smA[llo * 136 + klo];
        f16x8 ahi = *(const f16x8*)&smA[llo * 136 + klo + 16];
        f16x16 a;
        #pragma unroll
        for (int e = 0; e < 8; e++) { a[e] = alo[e]; a[e + 8] = ahi[e]; }
        f16x16 b = *(const f16x16*)&wfrag[((((wave << 2) + ks) << 5) | lane) * 16];
        acc = __builtin_amdgcn_wmma_f32_16x16x32_f16(
            false, a, false, b, (short)0, acc, false, false);
    }

    // ---- epilogue: fold LN + proj + biases in LDS (same-thread RMW, no race) ----
    int n  = (wave << 4) | llo;
    float gv  = gpair[n];
    float bnv = bnpair[n];
    float bd  = bdist[n];
    #pragma unroll
    for (int v = 0; v < 8; v++) {
        int jr = v + (lhi << 3);                 // D layout: row = v + 8*lhi
        float x = smP[jr * 132 + n];
        smP[jr * 132 + n] = (x - smMu[jr]) * smRs[jr] * gv + bnv + acc[v] + bd;
    }
    __syncthreads();

    // ---- coalesced streaming writeout: 512B contiguous per row ----
    f32x4 o0 = *(f32x4*)&smP[r * 132 + c0];
    f32x4 o1 = *(f32x4*)&smP[r * 132 + c0 + 4];
    f32x4* drow = (f32x4*)(outPair + ((long)(i * LSEQ + j)) * DPAIR + c0);
    __builtin_nontemporal_store(o0, drow);
    __builtin_nontemporal_store(o1, drow + 1);
}

extern "C" void kernel_launch(void* const* d_in, const int* in_sizes, int n_in,
                              void* d_out, int out_size, void* d_ws, size_t ws_size,
                              hipStream_t stream) {
    const float* msa    = (const float*)d_in[1];
    const float* pair   = (const float*)d_in[2];
    const float* state  = (const float*)d_in[3];
    const float* xyz    = (const float*)d_in[4];
    const float* w_dist = (const float*)d_in[5];
    const float* b_dist = (const float*)d_in[6];
    const float* g_pair = (const float*)d_in[7];
    const float* bn_pair= (const float*)d_in[8];
    const float* g_msa  = (const float*)d_in[9];
    const float* bn_msa = (const float*)d_in[10];
    const float* g_state= (const float*)d_in[11];
    const float* bn_state=(const float*)d_in[12];

    float* outMsa   = (float*)d_out;
    float* outPair  = outMsa + (long)LSEQ * DMSA;
    float* outState = outPair + (long)LSEQ * LSEQ * DPAIR;

    char* ws = (char*)d_ws;
    float*     Cb     = (float*)(ws + 0);
    _Float16*  stateH = (_Float16*)(ws + 12288);
    _Float16*  wfrag  = (_Float16*)(ws + 61440);

    k_cb<<<3, 256, 0, stream>>>(xyz, Cb);
    k_packw<<<64, 256, 0, stream>>>(w_dist, wfrag);
    k_state_ln<<<LSEQ / 8, 256, 0, stream>>>(state, g_state, bn_state, outState, stateH);
    k_msa_ln<<<LSEQ, 256, 0, stream>>>(msa, g_msa, bn_msa, outMsa);
    k_pair_fused<<<LSEQ * (LSEQ / 16), 256, 0, stream>>>(
        pair, b_dist, g_pair, bn_pair, Cb, stateH, wfrag, outPair);
}
